// Int4Linear_52544629899270
// MI455X (gfx1250) — matrix operand order repaired and checked by reading, not compiled, not run on hardware
//
#include <hip/hip_runtime.h>

typedef __attribute__((ext_vector_type(16))) _Float16 v16h;
typedef __attribute__((ext_vector_type(8)))  _Float16 v8h;
typedef __attribute__((ext_vector_type(8)))  float    v8f;
typedef __attribute__((ext_vector_type(4)))  float    v4f;
typedef __attribute__((ext_vector_type(4)))  int      v4i;

#define OUT_F 11008
#define IN_F  4096
#define MDIM  8192      // B*S
#define KB    32        // K-step per WMMA
#define MBLK  128
#define NBLK  128
#define LDSTR 40        // LDS row stride in halves (80B: 16B-aligned, bank-spread)

__launch_bounds__(256)
__global__ void int4_linear_wmma(const float* __restrict__ x,
                                 const int*   __restrict__ packed,
                                 const float* __restrict__ scale,
                                 const float* __restrict__ bias,
                                 float*       __restrict__ out)
{
    __shared__ __align__(16) _Float16 Ahi[MBLK * LDSTR];
    __shared__ __align__(16) _Float16 Alo[MBLK * LDSTR];
    __shared__ __align__(16) _Float16 Bs [NBLK * LDSTR];

    const int tid  = threadIdx.x;
    const int lane = tid & 31;
    const int wave = tid >> 5;       // 0..7
    const int wm   = wave >> 1;      // 0..3 : M direction
    const int wn   = wave & 1;       // 0..1 : N direction

    const int m_block = blockIdx.y * MBLK;
    const int n_block = blockIdx.x * NBLK;

    // ---- staging assignments (256 threads) ----
    // A: 128 rows x 32 floats ; thread -> (row, 16-float half)
    const int ar = tid >> 1;
    const int ah = tid & 1;
    // B: 128 rows x 16 int32 (32 nibbles) ; thread -> (row, 8-int32 half)
    const int br = tid >> 1;
    const int bh = tid & 1;

    const float* xrow = x      + (size_t)(m_block + ar) * IN_F      + ah * 16;
    const int*   prow = packed + (size_t)(n_block + br) * (IN_F/2)  + bh * 8;

    // accumulators: 2 (mt) x 4 (nt) tiles of 16x16 f32
    v8f acc[2][4];
    #pragma unroll
    for (int i = 0; i < 2; ++i)
        #pragma unroll
        for (int j = 0; j < 4; ++j)
            acc[i][j] = (v8f)(0.0f);

    // register prefetch buffers
    v4f a_g[4];
    v4i b_g[2];
    #pragma unroll
    for (int i = 0; i < 4; ++i) a_g[i] = *(const v4f*)(xrow + i * 4);
    #pragma unroll
    for (int i = 0; i < 2; ++i) b_g[i] = *(const v4i*)(prow + i * 4);

    const int NK = IN_F / KB;   // 128
    for (int kk = 0; kk < NK; ++kk) {
        __syncthreads();        // previous iteration done reading LDS

        // ---- stage A: fp32 -> (hi, lo) f16 split ----
        {
            _Float16 hbuf[16], lbuf[16];
            #pragma unroll
            for (int i = 0; i < 4; ++i) {
                #pragma unroll
                for (int j = 0; j < 4; ++j) {
                    float v = a_g[i][j];
                    _Float16 h = (_Float16)v;           // RNE
                    float r = v - (float)h;
                    hbuf[i * 4 + j] = h;
                    lbuf[i * 4 + j] = (_Float16)r;
                }
            }
            v8h* dh = (v8h*)&Ahi[ar * LDSTR + ah * 16];
            v8h* dl = (v8h*)&Alo[ar * LDSTR + ah * 16];
            dh[0] = *(const v8h*)&hbuf[0];  dh[1] = *(const v8h*)&hbuf[8];
            dl[0] = *(const v8h*)&lbuf[0];  dl[1] = *(const v8h*)&lbuf[8];
        }
        // ---- stage B: unpack int4 nibbles -> exact f16 integers ----
        {
            _Float16 qbuf[16];
            #pragma unroll
            for (int i = 0; i < 2; ++i) {
                #pragma unroll
                for (int j = 0; j < 4; ++j) {
                    int byt = b_g[i][j] & 0xFF;
                    int q0 = (byt & 15) - 7;           // low nibble  -> k=2j
                    int q1 = ((byt >> 4) & 15) - 7;    // high nibble -> k=2j+1
                    qbuf[(i * 4 + j) * 2 + 0] = (_Float16)q0;
                    qbuf[(i * 4 + j) * 2 + 1] = (_Float16)q1;
                }
            }
            v8h* db = (v8h*)&Bs[br * LDSTR + bh * 16];
            db[0] = *(const v8h*)&qbuf[0];  db[1] = *(const v8h*)&qbuf[8];
        }

        __syncthreads();        // LDS tile ready

        // prefetch next K-step from global (overlaps with WMMA below)
        if (kk + 1 < NK) {
            const float* xn = xrow + (size_t)(kk + 1) * KB;
            const int*   pn = prow + (size_t)(kk + 1) * (KB / 2);
            #pragma unroll
            for (int i = 0; i < 4; ++i) a_g[i] = *(const v4f*)(xn + i * 4);
            #pragma unroll
            for (int i = 0; i < 2; ++i) b_g[i] = *(const v4i*)(pn + i * 4);
        }

        // ---- build fragments per ISA 7.12.2 layouts ----
        const int mrow = lane & 15;
        const int kbA  = (lane < 16) ? 0 : 8;    // A: e0..7 -> K=kbA+e ; e8..15 -> K=kbA+16+(e-8)
        const int kbB  = (lane < 16) ? 0 : 8;    // B: e0..7 -> K=kbB+e ; e8..15 -> K=kbB+16+(e-8)

        v16h afh[2], afl[2], bf[4];
        #pragma unroll
        for (int mt = 0; mt < 2; ++mt) {
            const int m = wm * 32 + mt * 16 + mrow;
            v8h h0 = *(const v8h*)&Ahi[m * LDSTR + kbA];
            v8h h1 = *(const v8h*)&Ahi[m * LDSTR + kbA + 16];
            v8h l0 = *(const v8h*)&Alo[m * LDSTR + kbA];
            v8h l1 = *(const v8h*)&Alo[m * LDSTR + kbA + 16];
            afh[mt] = __builtin_shufflevector(h0, h1, 0,1,2,3,4,5,6,7,8,9,10,11,12,13,14,15);
            afl[mt] = __builtin_shufflevector(l0, l1, 0,1,2,3,4,5,6,7,8,9,10,11,12,13,14,15);
        }
        #pragma unroll
        for (int nt = 0; nt < 4; ++nt) {
            const int n = wn * 64 + nt * 16 + mrow;
            const v8h* pb = (const v8h*)&Bs[n * LDSTR + kbB];
            bf[nt] = __builtin_shufflevector(pb[0], pb[2], 0,1,2,3,4,5,6,7,8,9,10,11,12,13,14,15);
        }

        // ---- WMMA: hi and lo passes accumulate into same fp32 tile ----
        #pragma unroll
        for (int mt = 0; mt < 2; ++mt) {
            #pragma unroll
            for (int nt = 0; nt < 4; ++nt) {
                acc[mt][nt] = __builtin_amdgcn_wmma_f32_16x16x32_f16(
                    false, afh[mt], false, bf[nt], (short)0, acc[mt][nt], false, false);
                acc[mt][nt] = __builtin_amdgcn_wmma_f32_16x16x32_f16(
                    false, afl[mt], false, bf[nt], (short)0, acc[mt][nt], false, false);
            }
        }
    }

    // ---- epilogue: y = acc * scale[n] + bias[n] ----
    const int nlane = lane & 15;
    const int moff  = (lane < 16) ? 0 : 8;   // C/D layout: VGPR r -> row r + moff
    #pragma unroll
    for (int nt = 0; nt < 4; ++nt) {
        const int n  = n_block + wn * 64 + nt * 16 + nlane;
        const float sc = scale[n];
        const float bi = bias[n];
        #pragma unroll
        for (int mt = 0; mt < 2; ++mt) {
            const int mbase = m_block + wm * 32 + mt * 16 + moff;
            #pragma unroll
            for (int r = 0; r < 8; ++r) {
                out[(size_t)(mbase + r) * OUT_F + n] = acc[mt][nt][r] * sc + bi;
            }
        }
    }
}

extern "C" void kernel_launch(void* const* d_in, const int* in_sizes, int n_in,
                              void* d_out, int out_size, void* d_ws, size_t ws_size,
                              hipStream_t stream) {
    (void)in_sizes; (void)n_in; (void)out_size; (void)d_ws; (void)ws_size;
    const float* x      = (const float*)d_in[0];
    const int*   packed = (const int*)  d_in[1];
    const float* scale  = (const float*)d_in[2];
    const float* bias   = (const float*)d_in[3];
    float*       out    = (float*)d_out;

    dim3 grid(OUT_F / NBLK, MDIM / MBLK);   // 86 x 64 blocks
    int4_linear_wmma<<<grid, 256, 0, stream>>>(x, packed, scale, bias, out);
}
